// PathAgents_87943750353311
// MI455X (gfx1250) — matrix operand order repaired and checked by reading, not compiled, not run on hardware
//
#include <hip/hip_runtime.h>
#include <math.h>

#define B_    2
#define L_    1000
#define V_    8
#define DIN   1024
#define DM    512
#define DS    4
#define DCONV 4
#define DI    1024
#define DTR   32
#define DBC   (DTR + 2*DS)   /* 40 */
#define NL    2
#define NC    4
#define BL    (B_*L_)        /* 2000  */
#define VBL   (V_*BL)        /* 16000 */

typedef __attribute__((ext_vector_type(16))) __bf16        bf16x16;
typedef __attribute__((ext_vector_type(8)))  float         floatx8;
typedef __attribute__((ext_vector_type(4)))  unsigned int  uintx4;

union FragAB { uintx4 q[2]; bf16x16 v; };

#if __has_builtin(__builtin_amdgcn_global_load_async_to_lds_b128)
#define ASYNC_LDS 1
#else
#define ASYNC_LDS 0
#endif

#if ASYNC_LDS
// Exact parameter types leaked by the round-2 diagnostic:
// param0 = 'int __attribute__((vector_size(16))) __device__ *'  (global, AS1)
typedef int v4i_g __attribute__((vector_size(4 * sizeof(int))));
typedef v4i_g __attribute__((address_space(1))) *as1_v4i;
typedef v4i_g __attribute__((address_space(3))) *as3_v4i;
#endif

__device__ __forceinline__ unsigned short bf16_1(float a) {
  unsigned int u = __float_as_uint(a);
  u += 0x7FFFu + ((u >> 16) & 1u);
  return (unsigned short)(u >> 16);
}

__device__ __forceinline__ unsigned int pack_bf16x2(float a, float b) {
  unsigned int ua = __float_as_uint(a); ua += 0x7FFFu + ((ua >> 16) & 1u);
  unsigned int ub = __float_as_uint(b); ub += 0x7FFFu + ((ub >> 16) & 1u);
  return (ua >> 16) | (ub & 0xFFFF0000u);
}

// 16-byte global -> LDS copy: async tensor path if the toolchain has it.
__device__ __forceinline__ void copy16_g2l(const unsigned short* g, unsigned short* l) {
#if ASYNC_LDS
  __builtin_amdgcn_global_load_async_to_lds_b128((as1_v4i)g, (as3_v4i)l, 0, 0);
#else
  *(uintx4*)l = *(const uintx4*)g;
#endif
}

__device__ __forceinline__ void async_copy_fence() {
#if ASYNC_LDS
#if __has_builtin(__builtin_amdgcn_s_wait_asynccnt)
  __builtin_amdgcn_s_wait_asynccnt(0);
#else
  asm volatile("s_wait_asynccnt 0x0" ::: "memory");
#endif
#endif
}

__device__ __forceinline__ float block_sum_256(float v, float* sh) {
  int t = threadIdx.x;
  sh[t] = v;
  __syncthreads();
  for (int s = 128; s > 0; s >>= 1) {
    if (t < s) sh[t] += sh[t + s];
    __syncthreads();
  }
  float r = sh[0];
  __syncthreads();
  return r;
}

// ---------------------------------------------------------------------------
// Batched bf16 GEMM:  C[z](f32) = A[z] (MxK bf16) * W[z]^T (+bias), W = [N][K] bf16.
// 64x64 block tile, 4 wave32 (16 rows each x 4 n-subtiles), K stepped by 32.
// Staging via async global->LDS copies; v_wmma_f32_16x16x32_bf16 compute.
// ---------------------------------------------------------------------------
__global__ __launch_bounds__(128)
void wmma_gemm_bf16(const unsigned short* __restrict__ A, long long strideA, int lda,
                    const unsigned short* __restrict__ W, long long strideW, int ldw,
                    const float* __restrict__ bias, long long strideBias,
                    float* __restrict__ C, long long strideC, int ldc,
                    int M, int N, int K)
{
  __shared__ __align__(16) unsigned short As[64][40];
  __shared__ __align__(16) unsigned short Bs[64][40];

  const int tid  = threadIdx.x;
  const int lane = tid & 31;
  const int wave = tid >> 5;    // 0..3
  const int hi   = lane >> 4;   // 0/1
  const int lm   = lane & 15;

  const int nBase = blockIdx.x * 64;
  const int mBase = blockIdx.y * 64;
  const int z     = blockIdx.z;

  const unsigned short* Ab = A + (size_t)z * strideA;
  const unsigned short* Wb = W + (size_t)z * strideW;
  float*                Cb = C + (size_t)z * strideC;
  const float* biasb = bias ? (bias + (size_t)z * strideBias) : nullptr;

  const int mValid = (M - mBase < 64) ? (M - mBase) : 64;
  const int nValid = (N - nBase < 64) ? (N - nBase) : 64;

  // Zero-fill out-of-range tile rows ONCE; async loads never touch them.
  for (int q = tid; q < 64 * 4; q += 128) {
    int row = q >> 2, c8 = (q & 3) << 3;
    uintx4 zz = {0u, 0u, 0u, 0u};
    if (row >= mValid) *(uintx4*)&As[row][c8] = zz;
    if (row >= nValid) *(uintx4*)&Bs[row][c8] = zz;
  }

  floatx8 acc[4] = {};

  for (int k0 = 0; k0 < K; k0 += 32) {
    // Stage A tile (64 rows x 32 bf16 = 64B/row) and W tile, 16B chunks.
    for (int q = tid; q < 64 * 4; q += 128) {
      int row = q >> 2, c8 = (q & 3) << 3;
      if (row < mValid)
        copy16_g2l(Ab + (size_t)(mBase + row) * lda + k0 + c8, &As[row][c8]);
      if (row < nValid)
        copy16_g2l(Wb + (size_t)(nBase + row) * ldw + k0 + c8, &Bs[row][c8]);
    }
    async_copy_fence();
    __syncthreads();

    // A fragment: lane m=lm, elems 0..7 -> K=hi*8+0..7, 8..15 -> K=16+hi*8+0..7
    FragAB aF;
    aF.q[0] = *(const uintx4*)&As[wave * 16 + lm][hi * 8];
    aF.q[1] = *(const uintx4*)&As[wave * 16 + lm][16 + hi * 8];
    #pragma unroll
    for (int nt = 0; nt < 4; ++nt) {
      // B fragment: lane n=lm, elems 0..7 -> K=hi*8+0..7, 8..15 -> K=16+hi*8+0..7
      FragAB bF;
      bF.q[0] = *(const uintx4*)&Bs[nt * 16 + lm][hi * 8];
      bF.q[1] = *(const uintx4*)&Bs[nt * 16 + lm][16 + hi * 8];
      acc[nt] = __builtin_amdgcn_wmma_f32_16x16x32_bf16(
          false, aF.v, false, bF.v, (short)0, acc[nt], false, false);
    }
    __syncthreads();
  }

  // C/D layout: VGPR i, lane -> row = hi*8 + i, col = lm
  #pragma unroll
  for (int nt = 0; nt < 4; ++nt) {
    int gn = nBase + nt * 16 + lm;
    if (gn >= N) continue;
    float bv = biasb ? biasb[gn] : 0.f;
    #pragma unroll
    for (int i = 0; i < 8; ++i) {
      int gm = mBase + wave * 16 + hi * 8 + i;
      if (gm < M) Cb[(size_t)gm * ldc + gn] = acc[nt][i] + bv;
    }
  }
}

// ---------------------------------------------------------------------------
// f32 -> bf16 pair conversion (weights, once per launch)
// ---------------------------------------------------------------------------
__global__ void cvt_bf16_kernel(const float* __restrict__ src,
                                unsigned short* __restrict__ dst, int nPairs)
{
  int i = blockIdx.x * blockDim.x + threadIdx.x;
  if (i >= nPairs) return;
  ((unsigned int*)dst)[i] = pack_bf16x2(src[2 * i], src[2 * i + 1]);
}

// ---------------------------------------------------------------------------
// Batchnorm statistics over (B,L,V) per channel -> fused scale/shift
// ---------------------------------------------------------------------------
__global__ void bn_stats_kernel(const float* __restrict__ x,
                                const float* __restrict__ bn_w,
                                const float* __restrict__ bn_b,
                                float* __restrict__ scale,
                                float* __restrict__ shift)
{
  __shared__ float sh[256];
  int c = blockIdx.x;
  float a = 0.f, b = 0.f;
  for (int i = threadIdx.x; i < VBL; i += 256) {
    float v = x[(size_t)i * DIN + c];
    a += v; b += v * v;
  }
  float asum = block_sum_256(a, sh);
  float bsum = block_sum_256(b, sh);
  if (threadIdx.x == 0) {
    float mean = asum / (float)VBL;
    float var  = bsum / (float)VBL - mean * mean;
    float sc = bn_w[c] * rsqrtf(var + 1e-5f);
    scale[c] = sc;
    shift[c] = bn_b[c] - mean * sc;
  }
}

// ---------------------------------------------------------------------------
// Apply batchnorm + layout remap (b,l,v,c) -> row (b*V+v)*L+l, bf16 output
// ---------------------------------------------------------------------------
__global__ void bn_apply_bf16_kernel(const float* __restrict__ x,
                                     const float* __restrict__ scale,
                                     const float* __restrict__ shift,
                                     unsigned short* __restrict__ out)
{
  int idx = blockIdx.x * blockDim.x + threadIdx.x;  // bf16 pair index
  if (idx >= VBL * DIN / 2) return;
  int c   = (idx * 2) & (DIN - 1);
  int r   = (idx * 2) / DIN;        // (b*V+v)*L + l
  int rho = r / L_;
  int l   = r - rho * L_;
  int b   = rho / V_;
  int v   = rho - b * V_;
  const float* src = x + ((size_t)((b * L_ + l) * V_ + v)) * DIN + c;
  float f0 = src[0] * scale[c]     + shift[c];
  float f1 = src[1] * scale[c + 1] + shift[c + 1];
  ((unsigned int*)out)[idx] = pack_bf16x2(f0, f1);
}

// ---------------------------------------------------------------------------
// LayerNorm + exact GELU + sinusoidal PE; remap rows (b*V+v,l) -> (v*B+b,l)
// ---------------------------------------------------------------------------
__global__ void ln_gelu_pe_kernel(const float* __restrict__ in,
                                  const float* __restrict__ g,
                                  const float* __restrict__ bta,
                                  float* __restrict__ out)
{
  __shared__ float sh[256];
  int r   = blockIdx.x;           // (b*V+v)*L + l
  int rho = r / L_;
  int l   = r - rho * L_;
  int b   = rho / V_;
  int v   = rho - b * V_;
  const float* row = in + (size_t)r * DM;

  float s = 0.f;
  for (int d = threadIdx.x; d < DM; d += 256) s += row[d];
  float mu = block_sum_256(s, sh) / (float)DM;
  float s2 = 0.f;
  for (int d = threadIdx.x; d < DM; d += 256) { float t = row[d] - mu; s2 += t * t; }
  float var = block_sum_256(s2, sh) / (float)DM;
  float rs = rsqrtf(var + 1e-6f);

  float* orow = out + ((size_t)(v * B_ + b) * L_ + l) * DM;
  const float pos = (float)l;
  for (int d = threadIdx.x; d < DM; d += 256) {
    float xn = (row[d] - mu) * rs * g[d] + bta[d];
    float ge = 0.5f * xn * (1.f + erff(xn * 0.7071067811865476f));
    int   i2  = d & ~1;
    float div = expf((float)i2 * (-9.210340371976184f / (float)DM));
    float pe  = (d & 1) ? cosf(pos * div) : sinf(pos * div);
    orow[d] = ge + pe;
  }
}

// ---------------------------------------------------------------------------
// Per-view LayerNorm, bf16 output (feeds in_proj GEMM only)
// ---------------------------------------------------------------------------
__global__ void ln_rows_bf16_kernel(const float* __restrict__ in,
                                    const float* __restrict__ gamma,
                                    const float* __restrict__ beta,
                                    unsigned short* __restrict__ out)
{
  __shared__ float sh[256];
  int r = blockIdx.x;
  int v = r / BL;
  const float* row = in + (size_t)r * DM;
  const float* g  = gamma + (size_t)v * DM;
  const float* bt = beta  + (size_t)v * DM;

  float s = 0.f;
  for (int d = threadIdx.x; d < DM; d += 256) s += row[d];
  float mu = block_sum_256(s, sh) / (float)DM;
  float s2 = 0.f;
  for (int d = threadIdx.x; d < DM; d += 256) { float t = row[d] - mu; s2 += t * t; }
  float var = block_sum_256(s2, sh) / (float)DM;
  float rs = rsqrtf(var + 1e-6f);

  unsigned short* orow = out + (size_t)r * DM;
  for (int d = threadIdx.x; d < DM; d += 256)
    orow[d] = bf16_1((row[d] - mu) * rs * g[d] + bt[d]);
}

// ---------------------------------------------------------------------------
// Causal depthwise conv (DCONV=4) + bias + SiLU. Writes f32 (scan) + bf16 (GEMM).
// ---------------------------------------------------------------------------
__global__ void conv_silu_kernel(const float* __restrict__ xz,
                                 const float* __restrict__ cw,  // (V, DI, DCONV)
                                 const float* __restrict__ cb,  // (V, DI)
                                 float* __restrict__ xs,        // (V, BL, DI)
                                 unsigned short* __restrict__ xs16)
{
  int idx = blockIdx.x * blockDim.x + threadIdx.x;
  if (idx >= V_ * BL * DI) return;
  int d = idx & (DI - 1);
  int r = (idx / DI) % BL;
  int v = idx / (DI * BL);
  int b = r / L_;
  int l = r - b * L_;
  const float* w = cw + ((size_t)v * DI + d) * DCONV;
  float acc = cb[(size_t)v * DI + d];
  #pragma unroll
  for (int k = 0; k < DCONV; ++k) {
    int ls = l + k - (DCONV - 1);
    if (ls >= 0)
      acc += xz[((size_t)v * BL + (size_t)b * L_ + ls) * (2 * DI) + d] * w[k];
  }
  float sg = acc / (1.f + expf(-acc));
  xs[(size_t)idx]   = sg;
  xs16[(size_t)idx] = bf16_1(sg);
}

// dbc[:, 0:32] (stride 40) -> bf16 dt operand for dt_proj GEMM
__global__ void dt_cvt_kernel(const float* __restrict__ dbc,
                              unsigned short* __restrict__ dt16)
{
  int i = blockIdx.x * blockDim.x + threadIdx.x;
  if (i >= VBL * DTR) return;
  int r = i >> 5, k = i & (DTR - 1);
  dt16[i] = bf16_1(dbc[(size_t)r * DBC + k]);
}

__global__ void softplus_kernel(float* __restrict__ p, int n) {
  int i = blockIdx.x * blockDim.x + threadIdx.x;
  if (i < n) { float x = p[i]; p[i] = (x > 20.f) ? x : log1pf(expf(x)); }
}

// ---------------------------------------------------------------------------
// Selective scan: sequential over L, parallel over (v,b,d). Fuses +Dp*xs and
// the SiLU(z) gate; writes the gated y directly as bf16 (out_proj operand).
// ---------------------------------------------------------------------------
__global__ void scan_kernel(const float* __restrict__ Alog,   // (V, DI, DS)
                            const float* __restrict__ Dp,     // (V, DI)
                            const float* __restrict__ dbc,    // (V, BL, DBC)
                            const float* __restrict__ delta,  // (V, BL, DI)
                            const float* __restrict__ xs,     // (V, BL, DI)
                            const float* __restrict__ xz,     // (V, BL, 2*DI), z half
                            unsigned short* __restrict__ y16) // (V, BL, DI)
{
  int d  = blockIdx.x * blockDim.x + threadIdx.x;
  int vb = blockIdx.y;
  int v  = vb / B_;
  int b  = vb - v * B_;
  float Aa[DS];
  #pragma unroll
  for (int s = 0; s < DS; ++s)
    Aa[s] = -expf(Alog[((size_t)v * DI + d) * DS + s]);
  float Dd = Dp[(size_t)v * DI + d];
  float h[DS] = {0.f, 0.f, 0.f, 0.f};
  size_t rowBase = (size_t)v * BL + (size_t)b * L_;
  for (int l = 0; l < L_; ++l) {
    size_t row = rowBase + l;
    float dt = delta[row * DI + d];
    float xv = xs[row * DI + d];
    const float* bc = dbc + row * DBC;
    float y = 0.f;
    #pragma unroll
    for (int s = 0; s < DS; ++s) {
      h[s] = expf(dt * Aa[s]) * h[s] + dt * xv * bc[DTR + s];
      y += h[s] * bc[DTR + DS + s];
    }
    float z = xz[row * (2 * DI) + DI + d];
    float g = z / (1.f + expf(-z));
    y16[row * DI + d] = bf16_1((y + Dd * xv) * g);
  }
}

__global__ void view_mean_add_kernel(const float* __restrict__ h,
                                     float* __restrict__ xv)
{
  int idx = blockIdx.x * blockDim.x + threadIdx.x;
  if (idx >= BL * DM) return;
  float vals[V_]; float s = 0.f;
  #pragma unroll
  for (int v = 0; v < V_; ++v) { vals[v] = h[(size_t)v * BL * DM + idx]; s += vals[v]; }
  s *= (1.f / (float)V_);
  #pragma unroll
  for (int v = 0; v < V_; ++v) xv[(size_t)v * BL * DM + idx] = vals[v] + s;
}

// summary = mean over views, bf16 (feeds scores GEMM only)
__global__ void summary_bf16_kernel(const float* __restrict__ h,
                                    unsigned short* __restrict__ out)
{
  int idx = blockIdx.x * blockDim.x + threadIdx.x;
  if (idx >= BL * DM) return;
  float s = 0.f;
  #pragma unroll
  for (int v = 0; v < V_; ++v) s += h[(size_t)v * BL * DM + idx];
  out[idx] = bf16_1(s * (1.f / (float)V_));
}

__global__ void softmax_views_kernel(float* __restrict__ sc) {
  int r = blockIdx.x * blockDim.x + threadIdx.x;
  if (r >= BL) return;
  float* p = sc + (size_t)r * V_;
  float m = p[0];
  #pragma unroll
  for (int v = 1; v < V_; ++v) m = fmaxf(m, p[v]);
  float e[V_]; float s = 0.f;
  #pragma unroll
  for (int v = 0; v < V_; ++v) { e[v] = expf(p[v] - m); s += e[v]; }
  float inv = 1.f / s;
  #pragma unroll
  for (int v = 0; v < V_; ++v) p[v] = e[v] * inv;
}

__global__ void gather_kernel(const float* __restrict__ h,
                              const float* __restrict__ w,
                              float* __restrict__ out)
{
  int idx = blockIdx.x * blockDim.x + threadIdx.x;
  if (idx >= BL * DM) return;
  int r = idx / DM;
  float s = 0.f;
  #pragma unroll
  for (int v = 0; v < V_; ++v) s += h[(size_t)v * BL * DM + idx] * w[(size_t)r * V_ + v];
  out[idx] = s;
}

__global__ void pool_kernel(const float* __restrict__ g, float* __restrict__ pooled) {
  int idx = blockIdx.x * blockDim.x + threadIdx.x;
  if (idx >= B_ * DM) return;
  int b = idx / DM, d = idx - b * DM;
  float s = 0.f;
  for (int l = 0; l < L_; ++l) s += g[((size_t)b * L_ + l) * DM + d];
  pooled[idx] = s * (1.f / (float)L_);
}

__global__ void head_kernel(const float* __restrict__ pooled,
                            const float* __restrict__ nw, const float* __restrict__ nb,
                            const float* __restrict__ clsW, const float* __restrict__ clsb,
                            float* __restrict__ out)
{
  __shared__ float sh[256];
  __shared__ float lg[NC];
  int b = blockIdx.x;
  int t = threadIdx.x;
  const float* p = pooled + (size_t)b * DM;

  float s = 0.f;
  for (int d = t; d < DM; d += 256) s += p[d];
  float mu = block_sum_256(s, sh) / (float)DM;
  float s2 = 0.f;
  for (int d = t; d < DM; d += 256) { float u = p[d] - mu; s2 += u * u; }
  float var = block_sum_256(s2, sh) / (float)DM;
  float rs = rsqrtf(var + 1e-6f);
  for (int d = t; d < DM; d += 256)
    out[(size_t)b * DM + d] = (p[d] - mu) * rs * nw[d] + nb[d];

  for (int n = 0; n < NC; ++n) {
    float a = 0.f;
    for (int d = t; d < DM; d += 256) a += p[d] * clsW[(size_t)n * DM + d];
    float tot = block_sum_256(a, sh);
    if (t == 0) lg[n] = tot + clsb[n];
  }
  __syncthreads();
  if (t == 0) {
    float m = lg[0];
    for (int n = 1; n < NC; ++n) m = fmaxf(m, lg[n]);
    float e[NC]; float ssum = 0.f;
    for (int n = 0; n < NC; ++n) { e[n] = expf(lg[n] - m); ssum += e[n]; }
    for (int n = 0; n < NC; ++n) {
      out[B_ * DM + b * NC + n]           = lg[n];
      out[B_ * DM + B_ * NC + b * NC + n] = e[n] / ssum;
    }
  }
}

// ---------------------------------------------------------------------------
extern "C" void kernel_launch(void* const* d_in, const int* in_sizes, int n_in,
                              void* d_out, int out_size, void* d_ws, size_t ws_size,
                              hipStream_t stream)
{
  (void)in_sizes; (void)n_in; (void)out_size; (void)ws_size;
  const float* x         = (const float*)d_in[0];
  const float* bn_w      = (const float*)d_in[1];
  const float* bn_b      = (const float*)d_in[2];
  const float* fc_W      = (const float*)d_in[3];
  const float* fc_b      = (const float*)d_in[4];
  const float* emb_ln_w  = (const float*)d_in[5];
  const float* emb_ln_b  = (const float*)d_in[6];
  const float* m_ln_w    = (const float*)d_in[7];
  const float* m_ln_b    = (const float*)d_in[8];
  const float* in_proj_W = (const float*)d_in[9];
  const float* conv_w    = (const float*)d_in[10];
  const float* conv_b    = (const float*)d_in[11];
  const float* x_proj_W  = (const float*)d_in[12];
  const float* dt_proj_W = (const float*)d_in[13];
  const float* dt_proj_b = (const float*)d_in[14];
  const float* A_log     = (const float*)d_in[15];
  const float* Dp        = (const float*)d_in[16];
  const float* out_proj_W= (const float*)d_in[17];
  const float* gather_W  = (const float*)d_in[18];
  const float* gather_b  = (const float*)d_in[19];
  const float* norm_w    = (const float*)d_in[20];
  const float* norm_b    = (const float*)d_in[21];
  const float* cls_W     = (const float*)d_in[22];
  const float* cls_b     = (const float*)d_in[23];
  float* out = (float*)d_out;

  // ---- workspace carve-up (f32 region, then bf16 region) ----
  float* f = (float*)d_ws;
  float* w_scale = f; f += 1024;
  float* w_shift = f; f += 1024;
  float* bufA = f; f += (size_t)VBL * DM;        // fc out / h (f32)
  float* bufB = f; f += (size_t)VBL * DM;        // xv (f32)
  float* bufC = f; f += (size_t)VBL * 2 * DI;    // xz (f32; z half read by scan)
  float* bufD = f; f += (size_t)VBL * DI;        // xs (f32, scan input)
  float* bufE = f; f += (size_t)VBL * DBC;       // dbc (f32)
  float* bufF = f; f += (size_t)VBL * DI;        // delta (f32)
  float* bufG = f; f += (size_t)BL * DM;         // gathered (f32)
  float* bufS = f; f += (size_t)BL * V_;         // scores
  float* bufP = f; f += (size_t)B_ * DM;         // pooled
  unsigned short* u = (unsigned short*)f;
  unsigned short* xbn16 = u; u += (size_t)VBL * DIN;
  unsigned short* xln16 = u; u += (size_t)VBL * DM;
  unsigned short* xs16  = u; u += (size_t)VBL * DI;
  unsigned short* dt16  = u; u += (size_t)VBL * DTR;
  unsigned short* y16   = u; u += (size_t)VBL * DI;
  unsigned short* sum16 = u; u += (size_t)BL * DM;
  unsigned short* fcW16 = u; u += (size_t)DM * DIN;
  unsigned short* inW16 = u; u += (size_t)NL * V_ * 2 * DI * DM;
  unsigned short* xpW16 = u; u += (size_t)NL * V_ * DBC * DI;
  unsigned short* dtW16 = u; u += (size_t)NL * V_ * DI * DTR;
  unsigned short* opW16 = u; u += (size_t)NL * V_ * DM * DI;
  unsigned short* gaW16 = u; u += (size_t)V_ * DM;

  // ---- weight conversions (once per launch) ----
  {
    struct { const float* s; unsigned short* d; size_t n; } cv[6] = {
      { fc_W,      fcW16, (size_t)DM * DIN },
      { in_proj_W, inW16, (size_t)NL * V_ * 2 * DI * DM },
      { x_proj_W,  xpW16, (size_t)NL * V_ * DBC * DI },
      { dt_proj_W, dtW16, (size_t)NL * V_ * DI * DTR },
      { out_proj_W,opW16, (size_t)NL * V_ * DM * DI },
      { gather_W,  gaW16, (size_t)V_ * DM },
    };
    for (int i = 0; i < 6; ++i) {
      int pairs = (int)(cv[i].n / 2);
      cvt_bf16_kernel<<<(pairs + 255) / 256, 256, 0, stream>>>(cv[i].s, cv[i].d, pairs);
    }
  }

  // 1) batchnorm stats, then apply+remap to bf16
  bn_stats_kernel<<<DIN, 256, 0, stream>>>(x, bn_w, bn_b, w_scale, w_shift);
  {
    int pairs = VBL * DIN / 2;
    bn_apply_bf16_kernel<<<(pairs + 255) / 256, 256, 0, stream>>>(x, w_scale, w_shift, xbn16);
  }

  // 2) fc GEMM
  wmma_gemm_bf16<<<dim3(DM / 64, VBL / 64, 1), 128, 0, stream>>>(
      xbn16, 0, DIN, fcW16, 0, DIN, fc_b, 0, bufA, 0, DM, VBL, DM, DIN);

  // 3) LN + GELU + PE, remap to [v][b][l]
  ln_gelu_pe_kernel<<<VBL, 256, 0, stream>>>(bufA, emb_ln_w, emb_ln_b, bufB);

  for (int l = 0; l < NL; ++l) {
    ln_rows_bf16_kernel<<<VBL, 256, 0, stream>>>(
        bufB, m_ln_w + (size_t)l * V_ * DM, m_ln_b + (size_t)l * V_ * DM, xln16);

    wmma_gemm_bf16<<<dim3(2 * DI / 64, (BL + 63) / 64, V_), 128, 0, stream>>>(
        xln16, (long long)BL * DM, DM,
        inW16 + (size_t)l * V_ * 2 * DI * DM, (long long)2 * DI * DM, DM,
        nullptr, 0,
        bufC, (long long)BL * 2 * DI, 2 * DI,
        BL, 2 * DI, DM);

    {
      int n = V_ * BL * DI;
      conv_silu_kernel<<<(n + 255) / 256, 256, 0, stream>>>(
          bufC, conv_w + (size_t)l * V_ * DI * DCONV,
          conv_b + (size_t)l * V_ * DI, bufD, xs16);
    }

    wmma_gemm_bf16<<<dim3((DBC + 63) / 64, (BL + 63) / 64, V_), 128, 0, stream>>>(
        xs16, (long long)BL * DI, DI,
        xpW16 + (size_t)l * V_ * DBC * DI, (long long)DBC * DI, DI,
        nullptr, 0,
        bufE, (long long)BL * DBC, DBC,
        BL, DBC, DI);

    {
      int n = VBL * DTR;
      dt_cvt_kernel<<<(n + 255) / 256, 256, 0, stream>>>(bufE, dt16);
    }

    wmma_gemm_bf16<<<dim3(DI / 64, (BL + 63) / 64, V_), 128, 0, stream>>>(
        dt16, (long long)BL * DTR, DTR,
        dtW16 + (size_t)l * V_ * DI * DTR, (long long)DI * DTR, DTR,
        dt_proj_b + (size_t)l * V_ * DI, (long long)DI,
        bufF, (long long)BL * DI, DI,
        BL, DI, DTR);

    {
      int n = V_ * BL * DI;
      softplus_kernel<<<(n + 255) / 256, 256, 0, stream>>>(bufF, n);
    }

    scan_kernel<<<dim3(DI / 256, V_ * B_), 256, 0, stream>>>(
        A_log + (size_t)l * V_ * DI * DS, Dp + (size_t)l * V_ * DI,
        bufE, bufF, bufD, bufC, y16);

    wmma_gemm_bf16<<<dim3(DM / 64, (BL + 63) / 64, V_), 128, 0, stream>>>(
        y16, (long long)BL * DI, DI,
        opW16 + (size_t)l * V_ * DM * DI, (long long)DM * DI, DI,
        nullptr, 0,
        bufA, (long long)BL * DM, DM,
        BL, DM, DI);

    if (l < NL - 1) {
      int n = BL * DM;
      view_mean_add_kernel<<<(n + 255) / 256, 256, 0, stream>>>(bufA, bufB);
    }
  }

  // Final gather over views + head
  { int n = BL * DM; summary_bf16_kernel<<<(n + 255) / 256, 256, 0, stream>>>(bufA, sum16); }
  wmma_gemm_bf16<<<dim3(1, (BL + 63) / 64, 1), 128, 0, stream>>>(
      sum16, 0, DM, gaW16, 0, DM, gather_b, 0, bufS, 0, V_,
      BL, V_, DM);
  softmax_views_kernel<<<(BL + 255) / 256, 256, 0, stream>>>(bufS);
  { int n = BL * DM; gather_kernel<<<(n + 255) / 256, 256, 0, stream>>>(bufA, bufS, bufG); }
  { int n = B_ * DM; pool_kernel<<<(n + 255) / 256, 256, 0, stream>>>(bufG, bufP); }
  head_kernel<<<B_, 256, 0, stream>>>(bufP, norm_w, norm_b, cls_W, cls_b, out);
}
